// FSAS_26766236188756
// MI455X (gfx1250) — matrix operand order repaired and checked by reading, not compiled, not run on hardware
//
#include <hip/hip_runtime.h>

typedef __attribute__((ext_vector_type(16))) _Float16 v16h;
typedef __attribute__((ext_vector_type(8)))  _Float16 v8h;
typedef __attribute__((ext_vector_type(4)))  _Float16 v4h;
typedef __attribute__((ext_vector_type(8)))  float    v8f;
typedef __attribute__((ext_vector_type(4)))  float    v4f;

#define DIM   64
#define HID   384
#define NQK   128
#define HW    65536      // 256*256
#define W_IMG 256

// ---- LDS byte offsets (regions overlapped across pipeline phases) ----
#define OFF_HH    0        // _Float16 hh[384][100]  76800  (phase 2-3)
#define OFF_CONV  0        // float  conv[64][128]   32768  (phase 4-5, reuses hh)
#define OFF_ABUF  32768    // _Float16 aB[64][128]   16384  (phase 5-6, reuses hh)
#define OFF_WHID  76800    // _Float16 whid[384][64] 49152  (phase 0-2)
#define OFF_QS    76800    // _Float16 qs[128][64]   16384  (phase 3+, reuses whid)
#define OFF_KS    93184    // _Float16 ks[128][64]   16384
#define OFF_VS    109568   // _Float16 vs[64][128]   16384
#define OFF_XH    125952   // _Float16 xh[112][64]   14336  (phase 0-2)
#define OFF_BH    140288   // float bh[384]          1536
#define OFF_WDW   141824   // float wdw[384*9]       13824
#define OFF_BDW   155648   // float bdw[384]         1536
#define OFF_BPROJ 157184   // float bproj[64]        256
#define OFF_LNW   157440   // float lnw[128]         512
#define OFF_LNB   157952   // float lnb[128]         512
#define SMEM_SZ   158464

__global__ __launch_bounds__(256) void fsas_fused(
    const float* __restrict__ x,        const float* __restrict__ w_hidden,
    const float* __restrict__ b_hidden, const float* __restrict__ w_dw,
    const float* __restrict__ b_dw,     const float* __restrict__ w_proj,
    const float* __restrict__ b_proj,   const float* __restrict__ ln_w,
    const float* __restrict__ ln_b,     float* __restrict__ out)
{
  __shared__ __align__(16) unsigned char smem[SMEM_SZ];
  _Float16* hh    = (_Float16*)(smem + OFF_HH);
  float*    conv  = (float*)   (smem + OFF_CONV);
  _Float16* aB    = (_Float16*)(smem + OFF_ABUF);
  _Float16* whid  = (_Float16*)(smem + OFF_WHID);
  _Float16* qs    = (_Float16*)(smem + OFF_QS);
  _Float16* ksm   = (_Float16*)(smem + OFF_KS);
  _Float16* vs    = (_Float16*)(smem + OFF_VS);
  _Float16* xh    = (_Float16*)(smem + OFF_XH);
  float* bh_s     = (float*)(smem + OFF_BH);
  float* wdw_s    = (float*)(smem + OFF_WDW);
  float* bdw_s    = (float*)(smem + OFF_BDW);
  float* bproj_s  = (float*)(smem + OFF_BPROJ);
  float* lnw_s    = (float*)(smem + OFF_LNW);
  float* lnb_s    = (float*)(smem + OFF_LNB);

  const int tid = threadIdx.x;
  const int blk = blockIdx.x;            // 4096 = 4 batches * 32*32 patches
  const int b   = blk >> 10;
  const int pat = blk & 1023;
  const int py  = pat >> 5, px = pat & 31;
  const int y0  = py * 8,  x0 = px * 8;

  // ---------------- phase 0/1: stage weights + x halo (f32 -> f16) ----------
  for (int i = tid; i < HID*DIM; i += 256) whid[i] = (_Float16)w_hidden[i];
  for (int i = tid; i < HID;     i += 256) { bh_s[i] = b_hidden[i]; bdw_s[i] = b_dw[i]; }
  for (int i = tid; i < HID*9;   i += 256) wdw_s[i] = w_dw[i];
  if (tid < DIM) bproj_s[tid] = b_proj[tid];
  if (tid < NQK) { lnw_s[tid] = ln_w[tid]; lnb_s[tid] = ln_b[tid]; }

  const float* xb = x + (size_t)b * DIM * HW;
  for (int i = tid; i < DIM*100; i += 256) {        // 10x10 halo per channel
    int ch = i / 100, r = i % 100;
    int yy = y0 + r/10 - 1, xx = x0 + r%10 - 1;
    float v = 0.f;
    if ((unsigned)yy < 256u && (unsigned)xx < 256u)
      v = xb[(size_t)ch*HW + yy*W_IMG + xx];
    xh[r*DIM + ch] = (_Float16)v;                   // pixel-major (WMMA-A layout)
  }
  for (int i = tid; i < 12*DIM; i += 256)           // zero-pad rows 100..111
    xh[(100 + i/DIM)*DIM + (i % DIM)] = (_Float16)0.f;
  __syncthreads();

  const int wv  = tid >> 5;
  const int lm  = tid & 15;
  const int lhi = (tid >> 4) & 1;

  // ---------------- phase 2: GEMM1 via WMMA  (hidden = W_h @ x + b_h) -------
  // M=112 halo pixels (7 tiles), N=384 channels (24 tiles), K=64 (2 steps)
  for (int t = wv; t < 7*24; t += 8) {
    int mt = t % 7, nt = t / 7;
    const _Float16* arow = xh   + (mt*16 + lm)*DIM;
    const _Float16* brow = whid + (nt*16 + lm)*DIM;
    v8f acc = {};
    #pragma unroll
    for (int k0 = 0; k0 < DIM; k0 += 32) {
      union { v16h v; v8h h[2]; } ua;
      ua.h[0] = *(const v8h*)(arow + k0 + lhi*16);        // A: K 0-7 / 16-23
      ua.h[1] = *(const v8h*)(arow + k0 + lhi*16 + 8);    //    K 8-15 / 24-31
      v16h bv = *(const v16h*)(brow + k0 + lhi*16);       // B: K 0-15 / 16-31
      acc = __builtin_amdgcn_wmma_f32_16x16x32_f16(false, ua.v, false, bv,
                                                   (short)0, acc, false, false);
    }
    int   ch   = nt*16 + lm;                // D: lane holds 8 consecutive pixels of one ch
    int   p    = mt*16 + lhi*8;
    float bias = bh_s[ch];
    if (p + 4 <= 100) {
      v4h s0;
      #pragma unroll
      for (int r = 0; r < 4; ++r) s0[r] = (_Float16)(acc[r] + bias);
      *(v4h*)(hh + ch*100 + p) = s0;
    }
    if (p + 8 <= 100) {
      v4h s1;
      #pragma unroll
      for (int r = 0; r < 4; ++r) s1[r] = (_Float16)(acc[r+4] + bias);
      *(v4h*)(hh + ch*100 + p + 4) = s1;
    }
  }
  __syncthreads();

  // ---------------- phase 3: depthwise 3x3 + bias -> q, k, v ----------------
  for (int i = tid; i < HID*64; i += 256) {
    int ch = i >> 6, p = i & 63;
    int iy = p >> 3, ix = p & 7;
    const float*    wk   = wdw_s + ch*9;
    const _Float16* hrow = hh + ch*100;
    float acc = bdw_s[ch];
    #pragma unroll
    for (int ky = 0; ky < 3; ++ky)
      #pragma unroll
      for (int kx = 0; kx < 3; ++kx)
        acc += wk[ky*3+kx] * (float)hrow[(iy+ky)*10 + (ix+kx)];
    if      (ch < NQK)   qs [ch*64 + p]          = (_Float16)acc;  // channel-major
    else if (ch < 2*NQK) ksm[(ch-NQK)*64 + p]    = (_Float16)acc;  // channel-major
    else                 vs [p*NQK + (ch-2*NQK)] = (_Float16)acc;  // pixel-major
  }
  __syncthreads();

  // ---------------- phase 4: 8x8 patch circular convolution q (*) k ---------
  // out[i][j] = sum_{a,b} q[a][b] * k[(i-a)&7][(j-b)&7]
  for (int t = tid; t < NQK*8; t += 256) {
    int ch = t >> 3, oi = t & 7;
    const v8h* qrow = (const v8h*)(qs  + ch*64);
    const v8h* krow = (const v8h*)(ksm + ch*64);
    float outv[8] = {0,0,0,0,0,0,0,0};
    for (int a = 0; a < 8; ++a) {
      v8h qa = qrow[a];
      v8h kb = krow[(oi - a) & 7];
      float qf[8], kf[8];
      #pragma unroll
      for (int e = 0; e < 8; ++e) { qf[e] = (float)qa[e]; kf[e] = (float)kb[e]; }
      #pragma unroll
      for (int bq = 0; bq < 8; ++bq)
        #pragma unroll
        for (int j = 0; j < 8; ++j)
          outv[j] += qf[bq] * kf[(j-bq)&7];
    }
    #pragma unroll
    for (int j = 0; j < 8; ++j) conv[(oi*8+j)*NQK + ch] = outv[j];   // pixel-major
  }
  __syncthreads();

  // ---------------- phase 5: LayerNorm over 128 channels, multiply by v -----
  if (tid < 64) {
    const float* cp = conv + tid*NQK;
    float mu = 0.f;
    for (int c = 0; c < NQK; ++c) mu += cp[c];
    mu *= (1.f/NQK);
    float var = 0.f;
    for (int c = 0; c < NQK; ++c) { float d = cp[c]-mu; var += d*d; }
    var *= (1.f/NQK);
    float inv = rsqrtf(var + 1e-5f);
    for (int c = 0; c < NQK; ++c) {
      float nv = (cp[c]-mu)*inv*lnw_s[c] + lnb_s[c];
      aB[tid*NQK + c] = (_Float16)(nv * (float)vs[tid*NQK + c]);
    }
  }
  __syncthreads();

  // ---------------- phase 6: projection GEMM via WMMA -----------------------
  // out[p][o] = sum_c aB[p][c]*w_proj[o][c] + b_proj[o] ; M=64, N=64, K=128
  float* ob = out + (size_t)b * DIM * HW;
  for (int t = wv; t < 16; t += 8) {
    int mt = t & 3, nt = t >> 2;
    const _Float16* arow = aB + (mt*16 + lm)*NQK;
    const float*    brow = w_proj + (size_t)(nt*16 + lm)*NQK;   // L2-hot, cvt inline
    v8f acc = {};
    #pragma unroll
    for (int k0 = 0; k0 < NQK; k0 += 32) {
      union { v16h v; v8h h[2]; } ua;
      ua.h[0] = *(const v8h*)(arow + k0 + lhi*16);
      ua.h[1] = *(const v8h*)(arow + k0 + lhi*16 + 8);
      union { v16h v; _Float16 e[16]; } ub;
      const float* bp = brow + k0 + lhi*16;
      #pragma unroll
      for (int e = 0; e < 16; ++e) ub.e[e] = (_Float16)bp[e];
      acc = __builtin_amdgcn_wmma_f32_16x16x32_f16(false, ua.v, false, ub.v,
                                                   (short)0, acc, false, false);
    }
    int   o    = nt*16 + lm;
    float bias = bproj_s[o];
    int   prow = 2*mt + lhi;                       // patch row i; lane's 8 vals = j 0..7
    float* dst = ob + (size_t)o*HW + (size_t)(y0 + prow)*W_IMG + x0;
    v4f s0, s1;
    #pragma unroll
    for (int r = 0; r < 4; ++r) { s0[r] = acc[r] + bias; s1[r] = acc[r+4] + bias; }
    *(v4f*)dst       = s0;
    *(v4f*)(dst + 4) = s1;
  }
}

extern "C" void kernel_launch(void* const* d_in, const int* in_sizes, int n_in,
                              void* d_out, int out_size, void* d_ws, size_t ws_size,
                              hipStream_t stream) {
  (void)in_sizes; (void)n_in; (void)d_ws; (void)ws_size; (void)out_size;
  fsas_fused<<<dim3(4096), dim3(256), 0, stream>>>(
      (const float*)d_in[0], (const float*)d_in[1], (const float*)d_in[2],
      (const float*)d_in[3], (const float*)d_in[4], (const float*)d_in[5],
      (const float*)d_in[6], (const float*)d_in[7], (const float*)d_in[8],
      (float*)d_out);
}
